// SupConLoss_5634997093327
// MI455X (gfx1250) — matrix operand order repaired and checked by reading, not compiled, not run on hardware
//
#include <hip/hip_runtime.h>
#include <hip/hip_bf16.h>
#include <math.h>

#define BATCH 8192
#define DIM 256
#define TEMP 0.5f
#define INV_TEMP 2.0f
#define ROWMAX 2.0f            // analytic row max of logits = (1/T)*cos(self)=2
#define CSPLIT 2
#define TILES_TOTAL (BATCH/16)               // 512
#define TILES_PER_CHUNK (TILES_TOTAL/CSPLIT) // 256
#define WAVES_PER_WG 4
#define MAIN_BLOCK (WAVES_PER_WG*32)         // 128 threads = 4 waves (wave32)
#define MAIN_GRID_X (TILES_TOTAL/WAVES_PER_WG) // 128

#define LOG2E   1.4426950408889634f
#define NLOG2E2 (-2.0f*1.4426950408889634f)  // -2*log2(e): folds the -ROWMAX shift

typedef __attribute__((ext_vector_type(16))) _Float16 v16h;
typedef __attribute__((ext_vector_type(8)))  _Float16 v8h;
typedef __attribute__((ext_vector_type(8)))  float    v8f;

// -------- workspace layout (bytes) --------
// 0            : float accum[2] = {loss_sum, valid_count}
// 256          : float s_part [CSPLIT*BATCH]   (sum exp(logit-2), j != i)
// +CSPLIT*B*4  : float sp_part[CSPLIT*BATCH]   (sum of positive logits)
// +CSPLIT*B*4  : float np_part[CSPLIT*BATCH]   (count of positives)
// then         : _Float16 fh[BATCH*DIM]        (L2-normalized features, f16)
#define WS_ACCUM  0
#define WS_SPART  256
#define WS_SPPART (WS_SPART  + CSPLIT*BATCH*4)
#define WS_NPPART (WS_SPPART + CSPLIT*BATCH*4)
#define WS_FH     (WS_NPPART + CSPLIT*BATCH*4)

__global__ void k_init(float* __restrict__ accum) {
    if (threadIdx.x < 2) accum[threadIdx.x] = 0.0f;
}

// One block per row: L2-normalize and convert to f16.
__global__ __launch_bounds__(DIM) void k_norm(const float* __restrict__ f,
                                              _Float16* __restrict__ fh) {
    __shared__ float red[DIM];
    const int row = blockIdx.x, t = threadIdx.x;
    const float x = f[(size_t)row * DIM + t];
    red[t] = x * x;
    __syncthreads();
    #pragma unroll
    for (int s = DIM / 2; s > 0; s >>= 1) {
        if (t < s) red[t] += red[t + s];
        __syncthreads();
    }
    const float rn = rsqrtf(red[0]);
    fh[(size_t)row * DIM + t] = (_Float16)(x * rn);
}

// Fused GEMM (f fh^T via WMMA f16->f32) + SupCon per-row softmax partials.
// Each wave owns a 16-row strip; sweeps TILES_PER_CHUNK column tiles.
__global__ __launch_bounds__(MAIN_BLOCK) void k_main(
    const _Float16* __restrict__ fh, const int* __restrict__ labels,
    float* __restrict__ s_part, float* __restrict__ sp_part,
    float* __restrict__ np_part) {
    __shared__ int slbl[BATCH];                      // 32 KB of the 320 KB WGP LDS
    for (int i = threadIdx.x; i < BATCH; i += MAIN_BLOCK) slbl[i] = labels[i];
    __syncthreads();

    const int lane = threadIdx.x & 31;
    const int wave = threadIdx.x >> 5;
    const int n    = lane & 15;                      // column-within-tile / A row
    const int h    = lane >> 4;                      // lane half (K slicing)
    const int i0   = (blockIdx.x * WAVES_PER_WG + wave) * 16;  // strip base row
    const int cz   = blockIdx.y;                     // column chunk

    // ---- preload A fragments for all K (8 x 16x32 f16, contiguous K order) ----
    // lane (m + 16h) holds row m: K = 32k + 16h .. 16h+15
    union AV { v16h v; v8h h2[2]; };
    AV a[8];
    const _Float16* arow = fh + (size_t)(i0 + n) * DIM;
    #pragma unroll
    for (int kk = 0; kk < 8; ++kk) {
        a[kk].h2[0] = *(const v8h*)(arow + kk * 32 + 16 * h);
        a[kk].h2[1] = *(const v8h*)(arow + kk * 32 + 16 * h + 8);
    }

    int   rowlbl[8];
    float selfisf[8];                                // 1.0 on the diagonal lane slot
    #pragma unroll
    for (int r = 0; r < 8; ++r) {
        rowlbl[r]  = slbl[i0 + r + 8 * h];           // label of C-row r+8h
        selfisf[r] = (n == r + 8 * h) ? 1.0f : 0.0f;
    }

    float s[8], sp[8], np[8];
    #pragma unroll
    for (int r = 0; r < 8; ++r) { s[r] = 0.f; sp[r] = 0.f; np[r] = 0.f; }

    const int jt0 = cz * TILES_PER_CHUNK;
    for (int jt = jt0; jt < jt0 + TILES_PER_CHUNK; ++jt) {
        const int jbase = jt * 16;
        // B 32x16 f16 layout: lane (n + 16h) holds column n, K = 32k + 16h .. +15
        const _Float16* brow = fh + (size_t)(jbase + n) * DIM + 16 * h;

        const int lblj = slbl[jbase + n];            // issue ds_load early

        // Hoist all B fragments: one big load clause, drained across the WMMAs.
        v16h b[8];
        #pragma unroll
        for (int kk = 0; kk < 8; ++kk)
            b[kk] = *(const v16h*)(brow + kk * 32);

        v8f c = {};
        #pragma unroll
        for (int kk = 0; kk < 8; ++kk)
            c = __builtin_amdgcn_wmma_f32_16x16x32_f16(
                    false, a[kk].v, false, b[kk], (short)0, c, false, false);

        if (jt + 1 < jt0 + TILES_PER_CHUNK)
            __builtin_prefetch(brow + 16 * DIM, 0, 1);   // next B tile -> cache

        const float diagf = (jbase == i0) ? 1.0f : 0.0f; // uniform per wave
        #pragma unroll
        for (int r = 0; r < 8; ++r) {
            const float x       = c[r] * INV_TEMP;               // logit
            const float e       = exp2f(fmaf(x, LOG2E, NLOG2E2));// exp(x-2)
            const float notself = fmaf(-diagf, selfisf[r], 1.0f);// 0 iff self
            const float pos     = (lblj == rowlbl[r]) ? notself : 0.0f;
            s[r]  = fmaf(notself, e, s[r]);
            sp[r] = fmaf(pos, x, sp[r]);
            np[r] += pos;
        }
    }

    // ---- reduce across the 16 lanes of each C half (plain sums, no max merge) ----
    #pragma unroll
    for (int r = 0; r < 8; ++r) {
        #pragma unroll
        for (int mask = 1; mask < 16; mask <<= 1) {
            s[r]  += __shfl_xor(s[r],  mask, 32);
            sp[r] += __shfl_xor(sp[r], mask, 32);
            np[r] += __shfl_xor(np[r], mask, 32);
        }
    }
    if (n == 0) {
        #pragma unroll
        for (int r = 0; r < 8; ++r) {
            const int row = i0 + r + 8 * h;
            s_part [cz * BATCH + row] = s[r];
            sp_part[cz * BATCH + row] = sp[r];
            np_part[cz * BATCH + row] = np[r];
        }
    }
}

// Merge column chunks per row, compute per-row loss term, block-reduce, atomics.
__global__ __launch_bounds__(256) void k_merge(
    const float* __restrict__ s_part, const float* __restrict__ sp_part,
    const float* __restrict__ np_part, float* __restrict__ accum) {
    __shared__ float rl[256], rv[256];
    const int row = blockIdx.x * 256 + threadIdx.x;
    float s = 0.f, sp = 0.f, np = 0.f;
    #pragma unroll
    for (int c = 0; c < CSPLIT; ++c) {
        s  += s_part [c * BATCH + row];
        sp += sp_part[c * BATCH + row];
        np += np_part[c * BATCH + row];
    }
    const bool valid = np > 0.f;
    // -per_row = -(sp/np - M - log D) = M + log(s) - sp/np
    const float term = valid ? (ROWMAX + logf(s) - sp / np) : 0.f;
    rl[threadIdx.x] = term;
    rv[threadIdx.x] = valid ? 1.f : 0.f;
    __syncthreads();
    #pragma unroll
    for (int w = 128; w > 0; w >>= 1) {
        if (threadIdx.x < w) {
            rl[threadIdx.x] += rl[threadIdx.x + w];
            rv[threadIdx.x] += rv[threadIdx.x + w];
        }
        __syncthreads();
    }
    if (threadIdx.x == 0) {
        atomicAdd(&accum[0], rl[0]);
        atomicAdd(&accum[1], rv[0]);
    }
}

__global__ void k_final(const float* __restrict__ accum, float* __restrict__ out) {
    out[0] = TEMP * accum[0] / fmaxf(accum[1], 1.0f);   // SCALE_BY_TEMPERATURE
}

extern "C" void kernel_launch(void* const* d_in, const int* in_sizes, int n_in,
                              void* d_out, int out_size, void* d_ws, size_t ws_size,
                              hipStream_t stream) {
    const float* features = (const float*)d_in[0];
    const int*   labels   = (const int*)d_in[1];
    float*       out      = (float*)d_out;
    char*        ws       = (char*)d_ws;

    float*    accum   = (float*)(ws + WS_ACCUM);
    float*    s_part  = (float*)(ws + WS_SPART);
    float*    sp_part = (float*)(ws + WS_SPPART);
    float*    np_part = (float*)(ws + WS_NPPART);
    _Float16* fh      = (_Float16*)(ws + WS_FH);

    k_init<<<1, 32, 0, stream>>>(accum);
    k_norm<<<BATCH, DIM, 0, stream>>>(features, fh);

    dim3 grid(MAIN_GRID_X, CSPLIT);
    k_main<<<grid, MAIN_BLOCK, 0, stream>>>(fh, labels, s_part, sp_part, np_part);

    k_merge<<<BATCH / 256, 256, 0, stream>>>(s_part, sp_part, np_part, accum);
    k_final<<<1, 1, 0, stream>>>(accum, out);
}
